// LSTM_CRF_78099685310837
// MI455X (gfx1250) — compile-verified, matrix operand order checked by reading, not run on hardware
//
#include <hip/hip_runtime.h>
#include <cstddef>

// ---------- types ----------
typedef float  v8f   __attribute__((ext_vector_type(8)));
typedef __bf16 v16bf __attribute__((ext_vector_type(16)));
typedef __bf16 v8bf  __attribute__((ext_vector_type(8)));

#define B_   128
#define S_   256
#define E_   512
#define H_   1024
#define G4_  4096          // 4*H
#define KC_  1536          // E + H
#define TAUG_ 47
#define FSTR_ 48           // padded feat stride
#define START_ 45
#define STOP_  46

__device__ __forceinline__ v16bf cat16(v8bf lo, v8bf hi) {
    return __builtin_shufflevector(lo, hi, 0,1,2,3,4,5,6,7,8,9,10,11,12,13,14,15);
}

// ---------------- preprocessing ----------------
__global__ void k_cvt_bf16(const float* __restrict__ in, __bf16* __restrict__ out, int n) {
    int i = blockIdx.x * 256 + threadIdx.x;
    if (i < n) out[i] = (__bf16)in[i];
}

// Wcat[4096][1536] = [W_ih | W_hh] in bf16
__global__ void k_build_wcat(const float* __restrict__ Wih, const float* __restrict__ Whh,
                             __bf16* __restrict__ Wcat) {
    int i = blockIdx.x * 256 + threadIdx.x;           // < 4096*1536
    if (i >= G4_ * KC_) return;
    int r = i / KC_, k = i - r * KC_;
    float v = (k < E_) ? Wih[(size_t)r * E_ + k] : Whh[(size_t)r * H_ + (k - E_)];
    Wcat[i] = (__bf16)v;
}

// Woutp[64][1024] zero-padded bf16
__global__ void k_build_woutp(const float* __restrict__ Wout, __bf16* __restrict__ Wp) {
    int i = blockIdx.x * 256 + threadIdx.x;           // < 64*1024
    if (i >= 64 * H_) return;
    int r = i >> 10, k = i & 1023;
    Wp[i] = (__bf16)((r < TAUG_) ? Wout[(size_t)r * H_ + k] : 0.0f);
}

__global__ void k_init_state(float* __restrict__ C, __bf16* __restrict__ Hbf) {
    int i = blockIdx.x * 256 + threadIdx.x;           // < 128*1024
    if (i >= B_ * H_) return;
    C[i] = 0.0f;
    Hbf[i] = (__bf16)0.0f;
}

// ---------------- recurrent gate GEMM ----------------
// gates[128][4096] = [x_t ; h_{t-1}] (128 x 1536, bf16) @ Wcat^T, f32 accum.
// block = 256 threads (8 waves); wave w -> rows [16w,16w+16); block x -> cols [64*bx, +64)
__global__ void k_gemm_gates(const __bf16* __restrict__ Xbf,   // [B][S][E]
                             const __bf16* __restrict__ Hbf,   // [B][H]
                             const __bf16* __restrict__ Wcat,  // [4096][1536]
                             float* __restrict__ gates,        // [128][4096]
                             int t)
{
    const int lane = threadIdx.x & 31;
    const int wave = threadIdx.x >> 5;
    const int half = lane >> 4;           // K-half select per WMMA layout
    const int l16  = lane & 15;
    const int m0 = wave * 16;
    const int n0 = blockIdx.x * 64;
    const int mrow = m0 + l16;

    v8f acc0 = {}, acc1 = {}, acc2 = {}, acc3 = {};

    const __bf16* xrow = Xbf + ((size_t)mrow * S_ + t) * E_;
    const __bf16* hrow = Hbf + (size_t)mrow * H_;
    const __bf16* w0 = Wcat + (size_t)(n0 +  0 + l16) * KC_;
    const __bf16* w1 = Wcat + (size_t)(n0 + 16 + l16) * KC_;
    const __bf16* w2 = Wcat + (size_t)(n0 + 32 + l16) * KC_;
    const __bf16* w3 = Wcat + (size_t)(n0 + 48 + l16) * KC_;

    // ---- K over embed part [0,512) ----
    #pragma unroll 4
    for (int kc = 0; kc < E_; kc += 32) {
        const int k0 = kc + half * 8;
        v16bf a = cat16(*(const v8bf*)(xrow + k0), *(const v8bf*)(xrow + k0 + 16));
        v16bf b0 = cat16(*(const v8bf*)(w0 + k0), *(const v8bf*)(w0 + k0 + 16));
        v16bf b1 = cat16(*(const v8bf*)(w1 + k0), *(const v8bf*)(w1 + k0 + 16));
        v16bf b2 = cat16(*(const v8bf*)(w2 + k0), *(const v8bf*)(w2 + k0 + 16));
        v16bf b3 = cat16(*(const v8bf*)(w3 + k0), *(const v8bf*)(w3 + k0 + 16));
        acc0 = __builtin_amdgcn_wmma_f32_16x16x32_bf16(false, a, false, b0, (short)0, acc0, false, false);
        acc1 = __builtin_amdgcn_wmma_f32_16x16x32_bf16(false, a, false, b1, (short)0, acc1, false, false);
        acc2 = __builtin_amdgcn_wmma_f32_16x16x32_bf16(false, a, false, b2, (short)0, acc2, false, false);
        acc3 = __builtin_amdgcn_wmma_f32_16x16x32_bf16(false, a, false, b3, (short)0, acc3, false, false);
    }
    // ---- K over hidden part [512,1536) ----
    #pragma unroll 4
    for (int kc = 0; kc < H_; kc += 32) {
        const int k0 = kc + half * 8;
        const int w0k = E_ + k0;
        v16bf a = cat16(*(const v8bf*)(hrow + k0), *(const v8bf*)(hrow + k0 + 16));
        v16bf b0 = cat16(*(const v8bf*)(w0 + w0k), *(const v8bf*)(w0 + w0k + 16));
        v16bf b1 = cat16(*(const v8bf*)(w1 + w0k), *(const v8bf*)(w1 + w0k + 16));
        v16bf b2 = cat16(*(const v8bf*)(w2 + w0k), *(const v8bf*)(w2 + w0k + 16));
        v16bf b3 = cat16(*(const v8bf*)(w3 + w0k), *(const v8bf*)(w3 + w0k + 16));
        acc0 = __builtin_amdgcn_wmma_f32_16x16x32_bf16(false, a, false, b0, (short)0, acc0, false, false);
        acc1 = __builtin_amdgcn_wmma_f32_16x16x32_bf16(false, a, false, b1, (short)0, acc1, false, false);
        acc2 = __builtin_amdgcn_wmma_f32_16x16x32_bf16(false, a, false, b2, (short)0, acc2, false, false);
        acc3 = __builtin_amdgcn_wmma_f32_16x16x32_bf16(false, a, false, b3, (short)0, acc3, false, false);
    }

    // C/D layout: vgpr r, lane -> (m0 + r + 8*half, n0 + 16*j + l16)
    #pragma unroll
    for (int r = 0; r < 8; ++r) {
        const int m = m0 + r + 8 * half;
        float* gm = gates + (size_t)m * G4_ + n0 + l16;
        gm[ 0] = acc0[r];
        gm[16] = acc1[r];
        gm[32] = acc2[r];
        gm[48] = acc3[r];
    }
}

// ---------------- LSTM cell elementwise ----------------
__global__ void k_lstm_cell(const float* __restrict__ gates,
                            const float* __restrict__ b_ih, const float* __restrict__ b_hh,
                            float* __restrict__ C, __bf16* __restrict__ Hbf)
{
    int idx = blockIdx.x * 256 + threadIdx.x;         // < 128*1024
    if (idx >= B_ * H_) return;
    int m = idx >> 10, j = idx & 1023;
    const float* g = gates + (size_t)m * G4_;
    float gi = g[j]            + b_ih[j]            + b_hh[j];
    float gf = g[H_ + j]       + b_ih[H_ + j]       + b_hh[H_ + j];
    float gg = g[2 * H_ + j]   + b_ih[2 * H_ + j]   + b_hh[2 * H_ + j];
    float go = g[3 * H_ + j]   + b_ih[3 * H_ + j]   + b_hh[3 * H_ + j];
    float si = 1.0f / (1.0f + expf(-gi));
    float sf = 1.0f / (1.0f + expf(-gf));
    float so = 1.0f / (1.0f + expf(-go));
    float c = sf * C[idx] + si * tanhf(gg);
    C[idx] = c;
    float h = so * tanhf(c);
    Hbf[idx] = (__bf16)h;
}

// ---------------- hidden->tag + SELU ----------------
// grid = 4 (N tiles of 16), block = 256 (8 waves = 8 M tiles), K = 1024
__global__ void k_feats(const __bf16* __restrict__ Hbf, const __bf16* __restrict__ Wp,
                        const float* __restrict__ b_out, float* __restrict__ Feat, int t)
{
    const int lane = threadIdx.x & 31;
    const int wave = threadIdx.x >> 5;
    const int half = lane >> 4;
    const int l16  = lane & 15;
    const int m0 = wave * 16;
    const int n0 = blockIdx.x * 16;

    const __bf16* arow = Hbf + (size_t)(m0 + l16) * H_;
    const __bf16* brow = Wp  + (size_t)(n0 + l16) * H_;
    v8f acc = {};
    #pragma unroll 4
    for (int kc = 0; kc < H_; kc += 32) {
        const int k0 = kc + half * 8;
        v16bf a = cat16(*(const v8bf*)(arow + k0), *(const v8bf*)(arow + k0 + 16));
        v16bf b = cat16(*(const v8bf*)(brow + k0), *(const v8bf*)(brow + k0 + 16));
        acc = __builtin_amdgcn_wmma_f32_16x16x32_bf16(false, a, false, b, (short)0, acc, false, false);
    }
    const int n = n0 + l16;
    if (n < TAUG_) {
        const float bo = b_out[n];
        const float alpha = 1.6732632423543772f, scale = 1.0507009873554805f;
        #pragma unroll
        for (int r = 0; r < 8; ++r) {
            const int m = m0 + r + 8 * half;
            float x = acc[r] + bo;
            float s = (x > 0.0f) ? x : alpha * (expf(x) - 1.0f);
            Feat[((size_t)m * S_ + t) * FSTR_ + n] = scale * s;
        }
    }
}

// ---------------- CRF log-partition (per batch element) ----------------
__global__ void k_crf(const float* __restrict__ Feat, const float* __restrict__ trans,
                      float* __restrict__ out)
{
    __shared__ float Ts[TAUG_ * FSTR_];
    __shared__ float alpha[FSTR_];
    const int b = blockIdx.x, tid = threadIdx.x;      // 64 threads

    for (int idx = tid; idx < TAUG_ * TAUG_; idx += 64) {
        int i = idx / TAUG_, j = idx - i * TAUG_;
        Ts[i * FSTR_ + j] = trans[idx];
    }
    if (tid < TAUG_) alpha[tid] = (tid == START_) ? 0.0f : -10000.0f;
    __syncthreads();

    const float* fb = Feat + (size_t)b * S_ * FSTR_;
    for (int t = 0; t < S_; ++t) {
        float r = 0.0f;
        if (tid < TAUG_) {
            const float* Tj = Ts + tid * FSTR_;
            float m = -3.4e38f;
            #pragma unroll 1
            for (int i = 0; i < TAUG_; ++i) m = fmaxf(m, alpha[i] + Tj[i]);
            float s = 0.0f;
            #pragma unroll 1
            for (int i = 0; i < TAUG_; ++i) s += expf(alpha[i] + Tj[i] - m);
            r = fb[t * FSTR_ + tid] + m + logf(s);
        }
        __syncthreads();
        if (tid < TAUG_) alpha[tid] = r;
        __syncthreads();
    }
    if (tid == 0) {
        const float* Tj = Ts + STOP_ * FSTR_;
        float m = -3.4e38f;
        for (int i = 0; i < TAUG_; ++i) m = fmaxf(m, alpha[i] + Tj[i]);
        float s = 0.0f;
        for (int i = 0; i < TAUG_; ++i) s += expf(alpha[i] + Tj[i] - m);
        out[b] = m + logf(s);
    }
}

// ---------------- launcher ----------------
extern "C" void kernel_launch(void* const* d_in, const int* in_sizes, int n_in,
                              void* d_out, int out_size, void* d_ws, size_t ws_size,
                              hipStream_t stream)
{
    const float* embeds = (const float*)d_in[0];   // (128,256,512)
    const float* W_ih   = (const float*)d_in[1];   // (4096,512)
    const float* W_hh   = (const float*)d_in[2];   // (4096,1024)
    const float* b_ih   = (const float*)d_in[3];   // (4096,)
    const float* b_hh   = (const float*)d_in[4];   // (4096,)
    const float* W_out  = (const float*)d_in[5];   // (47,1024)
    const float* b_out  = (const float*)d_in[6];   // (47,)
    const float* trans  = (const float*)d_in[7];   // (47,47)
    float* out = (float*)d_out;                    // (128,)

    char* ws = (char*)d_ws;
    size_t off = 0;
    float*  gates = (float*) (ws + off); off += (size_t)B_ * G4_ * 4;          //  2.00 MB
    __bf16* Wcat  = (__bf16*)(ws + off); off += (size_t)G4_ * KC_ * 2;         // 12.58 MB
    __bf16* Xbf   = (__bf16*)(ws + off); off += (size_t)B_ * S_ * E_ * 2;      // 33.55 MB
    __bf16* Woutp = (__bf16*)(ws + off); off += (size_t)64 * H_ * 2;           //  0.13 MB
    __bf16* Hbf   = (__bf16*)(ws + off); off += (size_t)B_ * H_ * 2;           //  0.26 MB
    float*  Cst   = (float*) (ws + off); off += (size_t)B_ * H_ * 4;           //  0.52 MB
    float*  Feat  = (float*) (ws + off); off += (size_t)B_ * S_ * FSTR_ * 4;   //  6.29 MB

    // preprocessing
    {
        int n = B_ * S_ * E_;
        k_cvt_bf16<<<(n + 255) / 256, 256, 0, stream>>>(embeds, Xbf, n);
        int nw = G4_ * KC_;
        k_build_wcat<<<(nw + 255) / 256, 256, 0, stream>>>(W_ih, W_hh, Wcat);
        k_build_woutp<<<(64 * H_ + 255) / 256, 256, 0, stream>>>(W_out, Woutp);
        k_init_state<<<(B_ * H_ + 255) / 256, 256, 0, stream>>>(Cst, Hbf);
    }

    // sequential recurrence: per-step launches (stream-ordered)
    for (int t = 0; t < S_; ++t) {
        k_gemm_gates<<<G4_ / 64, 256, 0, stream>>>(Xbf, Hbf, Wcat, gates, t);
        k_lstm_cell<<<(B_ * H_) / 256, 256, 0, stream>>>(gates, b_ih, b_hh, Cst, Hbf);
        k_feats<<<4, 256, 0, stream>>>(Hbf, Woutp, b_out, Feat, t);
    }

    // CRF chain, one block per batch element
    k_crf<<<B_, 64, 0, stream>>>(Feat, trans, out);
    (void)in_sizes; (void)n_in; (void)out_size; (void)ws_size;
}
